// VectorQuantizer_69690139345403
// MI455X (gfx1250) — matrix operand order repaired and checked by reading, not compiled, not run on hardware
//
#include <hip/hip_runtime.h>

// ---------------------------------------------------------------------------
// VQ quantizer for MI455X (gfx1250, wave32, WMMA).
//   x:        [32, 256, 32, 32] f32   (B, C, H, W), C == embedding dim
//   codebook: [8192, 256] f32
//   out:      xq [32,256,32,32] f32  ++  loss scalar  (8388609 floats)
//
// GEMM M=32768,N=8192,K=256 fused with row argmin.
// bf16x3 split precision on v_wmma_f32_16x16x32_bf16 (~2^-16 rel error).
// Persistent A fragments in VGPRs; codebook staged to LDS by async DMA
// (global_load_async_to_lds_b128, double buffered, s_wait_asynccnt).
// ---------------------------------------------------------------------------

typedef __attribute__((ext_vector_type(8)))  float  v8f;
typedef __attribute__((ext_vector_type(8)))  __bf16 v8bf;
typedef __attribute__((ext_vector_type(16))) __bf16 v16bf;
typedef __attribute__((__vector_size__(16))) int    v4i;   // async builtin operand type

#define C_DIM   256
#define K_DIM   8192
#define N_TOT   32768
#define BROWS   128          // pixel rows per block
#define KBUF    32           // codes per LDS buffer (2 tiles of 16 per wave)
#define CPAD    264          // LDS row stride in bf16 (528 B -> conflict-free b128 frag loads)
#define BUFELEMS (64 * CPAD) // one buffer: 32 rows hi + 32 rows lo
#define NUMEL   8388608      // 32*256*32*32

#define SHUF16 0,1,2,3,4,5,6,7,8,9,10,11,12,13,14,15

#if __has_builtin(__builtin_amdgcn_global_load_async_to_lds_b128) && \
    __has_builtin(__builtin_amdgcn_s_wait_asynccnt)
#define HAVE_ASYNC 1
#else
#define HAVE_ASYNC 0
#endif

// ------------- phase 0: ||e||^2 per code + bf16 hi/lo split of codebook -----
__global__ __launch_bounds__(256)
void vq_prep_kernel(const float* __restrict__ cb, float* __restrict__ enorm,
                    __bf16* __restrict__ cbhi, __bf16* __restrict__ cblo) {
  const int lane = threadIdx.x & 31;
  const int code = blockIdx.x * 8 + (threadIdx.x >> 5);
  const float* row = cb + (size_t)code * C_DIM;
  float s = 0.f;
  #pragma unroll
  for (int c = lane; c < C_DIM; c += 32) {
    float v = row[c];
    __bf16 hi = (__bf16)v;
    __bf16 lo = (__bf16)(v - (float)hi);
    cbhi[(size_t)code * C_DIM + c] = hi;
    cblo[(size_t)code * C_DIM + c] = lo;
    s += v * v;
  }
  #pragma unroll
  for (int m = 16; m >= 1; m >>= 1) s += __shfl_xor(s, m, 32);
  if (lane == 0) enorm[code] = s;
}

// ---------------- phase 1: fused GEMM (bf16x3 WMMA) + argmin ----------------
__global__ __launch_bounds__(256, 1)
void vq_argmin_kernel(const float* __restrict__ x,
                      const __bf16* __restrict__ cbhi, const __bf16* __restrict__ cblo,
                      const float* __restrict__ enorm, int* __restrict__ amin) {
  extern __shared__ char smem[];
  __bf16* s_bb  = (__bf16*)smem;                 // 2 x [ chi[32][CPAD] clo[32][CPAD] ]
  __bf16* s_xhi = s_bb + 2 * BUFELEMS;           // [BROWS][CPAD]
  __bf16* s_xlo = s_xhi + BROWS * CPAD;          // [BROWS][CPAD]

  const int tid  = threadIdx.x;
  const int lane = tid & 31;
  const int wave = tid >> 5;                     // 8 waves, 16 rows each
  const int rowBase = blockIdx.x * BROWS;
  const int bimg = rowBase >> 10;                // 1024 pixels per image
  const int hw0  = rowBase & 1023;

  const int hf    = lane >> 4;                   // lane half (0/1)
  const int l15   = lane & 15;
  const int kSelA = hf * 8;                      // A frag: K 0-7/16-23 vs 8-15/24-31
  const int bK    = hf * 8;                      // B frag: K 0-7/16-23 vs 8-15/24-31

  // Issue a double-buffer fill: 32 codes x 256 ch x {hi,lo} = 2048 x 16B chunks.
  auto fill_buf = [&](int bufIdx, int k0n) {
    __bf16* dstBase = s_bb + bufIdx * BUFELEMS;
    #pragma unroll
    for (int r = 0; r < 8; ++r) {
      int q     = tid + r * 256;                 // 0..2047
      int plane = q >> 10;                       // 0=hi, 1=lo
      int rem   = q & 1023;
      int kr    = rem >> 5;                      // code row 0..31
      int qc    = rem & 31;                      // 16B chunk in row
      const __bf16* g = (plane ? cblo : cbhi) + (size_t)(k0n + kr) * C_DIM + qc * 8;
      __bf16* l = dstBase + plane * (32 * CPAD) + kr * CPAD + qc * 8;
#if HAVE_ASYNC
      __builtin_amdgcn_global_load_async_to_lds_b128(
          (__attribute__((address_space(1))) v4i*)g,
          (__attribute__((address_space(3))) v4i*)l, 0, 0);
#else
      *(v8bf*)l = *(const v8bf*)g;
#endif
    }
  };

  fill_buf(0, 0);                                // prefetch first codes, overlaps x staging

  // Stage x tile: x[b, c, hw0+j] -> LDS row j, col c (split hi/lo bf16).
  // For fixed c the 128 j's are contiguous floats -> fully coalesced.
  {
    const float* xb = x + (size_t)bimg * (C_DIM * 1024) + hw0;
    const int j = tid & 127;
    for (int c = tid >> 7; c < C_DIM; c += 2) {
      float v = xb[(size_t)c * 1024 + j];
      __bf16 hi = (__bf16)v;
      __bf16 lo = (__bf16)(v - (float)hi);
      s_xhi[j * CPAD + c] = hi;
      s_xlo[j * CPAD + c] = lo;
    }
  }
#if HAVE_ASYNC
  __builtin_amdgcn_s_wait_asynccnt(0);
#endif
  __syncthreads();

  // Persistent A fragments for this wave's 16 rows (hi+lo, all of C): 128 VGPRs.
  v16bf aH[8], aL[8];
  {
    const __bf16* pAhi = s_xhi + (wave * 16 + l15) * CPAD;
    const __bf16* pAlo = s_xlo + (wave * 16 + l15) * CPAD;
    #pragma unroll
    for (int i = 0; i < 8; ++i) {
      int cc = i * 32;
      v8bf t0 = *(const v8bf*)(pAhi + cc + kSelA);
      v8bf t1 = *(const v8bf*)(pAhi + cc + 16 + kSelA);
      aH[i] = __builtin_shufflevector(t0, t1, SHUF16);
      v8bf t2 = *(const v8bf*)(pAlo + cc + kSelA);
      v8bf t3 = *(const v8bf*)(pAlo + cc + 16 + kSelA);
      aL[i] = __builtin_shufflevector(t2, t3, SHUF16);
    }
  }

  float mval[8];
  int   midx[8];
  #pragma unroll
  for (int r = 0; r < 8; ++r) { mval[r] = 3.4e38f; midx[r] = 0; }

  // One 16x16 tile: B streamed from LDS (only traffic in steady state).
  auto do_tile = [&](const __bf16* pBh, const __bf16* pBl, int codeBase) {
    v8f acc = {0.f, 0.f, 0.f, 0.f, 0.f, 0.f, 0.f, 0.f};
    #pragma unroll
    for (int i = 0; i < 8; ++i) {
      int cc = i * 32;
      v8bf b0 = *(const v8bf*)(pBh + cc + bK);
      v8bf b1 = *(const v8bf*)(pBh + cc + bK + 16);
      v16bf bh = __builtin_shufflevector(b0, b1, SHUF16);
      v8bf b2 = *(const v8bf*)(pBl + cc + bK);
      v8bf b3 = *(const v8bf*)(pBl + cc + bK + 16);
      v16bf bl = __builtin_shufflevector(b2, b3, SHUF16);
      // bf16x3: hi*hi + hi*lo + lo*hi, f32 accumulate
      acc = __builtin_amdgcn_wmma_f32_16x16x32_bf16(false, aH[i], false, bh, (short)0, acc, false, false);
      acc = __builtin_amdgcn_wmma_f32_16x16x32_bf16(false, aH[i], false, bl, (short)0, acc, false, false);
      acc = __builtin_amdgcn_wmma_f32_16x16x32_bf16(false, aL[i], false, bh, (short)0, acc, false, false);
    }
    // score = ||e||^2 - 2*sim  (||x||^2 constant per row)
    const int   code = codeBase + l15;
    const float en   = enorm[code];
    #pragma unroll
    for (int r = 0; r < 8; ++r) {
      float sc = en - 2.0f * acc[r];
      if (sc < mval[r]) { mval[r] = sc; midx[r] = code; }
    }
  };

  fill_buf(1, KBUF);                             // second buffer in flight

  const int NITER = K_DIM / KBUF;                // 256
  for (int kb = 0; kb < NITER; ++kb) {
    const int cur = kb & 1;
    const __bf16* bb  = s_bb + cur * BUFELEMS;
    const __bf16* ph0 = bb + l15 * CPAD;
    const __bf16* pl0 = bb + (32 + l15) * CPAD;
    do_tile(ph0,             pl0,             kb * KBUF);
    do_tile(ph0 + 16 * CPAD, pl0 + 16 * CPAD, kb * KBUF + 16);
#if HAVE_ASYNC
    __builtin_amdgcn_s_wait_asynccnt(0);         // next buffer's fill complete
#endif
    __syncthreads();                             // all waves done reading buf[cur]
    if (kb + 2 < NITER) fill_buf(cur, (kb + 2) * KBUF);
  }

  // D layout: VGPR r, lane half hf -> row r + 8*hf, col l15.
  // Min-reduce across the 16 lanes of each half (xor masks keep bit4 fixed).
  #pragma unroll
  for (int r = 0; r < 8; ++r) {
    float v = mval[r]; int ix = midx[r];
    #pragma unroll
    for (int m = 8; m >= 1; m >>= 1) {
      float ov = __shfl_xor(v, m, 32);
      int   oi = __shfl_xor(ix, m, 32);
      if (ov < v || (ov == v && oi < ix)) { v = ov; ix = oi; }  // first-index tiebreak
    }
    if (l15 == 0) amin[rowBase + wave * 16 + r + 8 * hf] = ix;
  }
}

// ------------- phase 2: gather xq, write output, partial MSE sums -----------
__global__ __launch_bounds__(256)
void vq_gather_loss_kernel(const float* __restrict__ x, const float* __restrict__ cb,
                           const int* __restrict__ amin, float* __restrict__ out,
                           float* __restrict__ partials) {
  float lsum = 0.f;
  const long long total  = NUMEL;
  const long long stride = (long long)gridDim.x * 256;
  for (long long e = (long long)blockIdx.x * 256 + threadIdx.x; e < total; e += stride) {
    int bimg = (int)(e >> 18);          // 262144 elems per image
    int c    = (int)((e >> 10) & 255);
    int hw   = (int)(e & 1023);
    int n    = bimg * 1024 + hw;        // flat pixel index (matches reference)
    int idx  = amin[n];
    float q  = cb[(size_t)idx * C_DIM + c];
    float xv = x[e];                    // out index == x index (both NCHW)
    out[e] = q;                         // straight-through: xq_st == xq numerically
    float d = q - xv;
    lsum += d * d;
  }
  __shared__ float red[256];
  red[threadIdx.x] = lsum;
  __syncthreads();
  #pragma unroll
  for (int s = 128; s >= 1; s >>= 1) {
    if (threadIdx.x < s) red[threadIdx.x] += red[threadIdx.x + s];
    __syncthreads();
  }
  if (threadIdx.x == 0) partials[blockIdx.x] = red[0];
}

// ----------------- phase 3: deterministic final loss reduction --------------
__global__ __launch_bounds__(256)
void vq_loss_final_kernel(const float* __restrict__ partials, int nparts,
                          float* __restrict__ loss_out) {
  __shared__ float red[256];
  float s = 0.f;
  for (int i = threadIdx.x; i < nparts; i += 256) s += partials[i];
  red[threadIdx.x] = s;
  __syncthreads();
  #pragma unroll
  for (int t = 128; t >= 1; t >>= 1) {
    if (threadIdx.x < t) red[threadIdx.x] += red[threadIdx.x + t];
    __syncthreads();
  }
  // loss = mean + BETA*mean = 1.25 * MSE
  if (threadIdx.x == 0) loss_out[0] = 1.25f * red[0] / (float)NUMEL;
}

// ---------------------------------------------------------------------------
extern "C" void kernel_launch(void* const* d_in, const int* in_sizes, int n_in,
                              void* d_out, int out_size, void* d_ws, size_t ws_size,
                              hipStream_t stream) {
  const float* x  = (const float*)d_in[0];   // [32,256,32,32]
  const float* cb = (const float*)d_in[1];   // [8192,256]
  float* out  = (float*)d_out;               // xq flat ++ loss
  float* loss = out + NUMEL;

  // bf16 hi/lo codebook planes stashed at the front of d_out (8.4 MB of 33.5 MB);
  // phase 2 fully overwrites d_out afterwards, so this is dead by validation time.
  __bf16* cbhi = (__bf16*)d_out;
  __bf16* cblo = cbhi + (size_t)K_DIM * C_DIM;

  // workspace: enorm[8192] f32 | amin[32768] i32 | partials[2048] f32  (~168 KB)
  float* enorm    = (float*)d_ws;
  int*   amin     = (int*)((char*)d_ws + (size_t)K_DIM * 4);
  float* partials = (float*)((char*)d_ws + (size_t)K_DIM * 4 + (size_t)N_TOT * 4);

  vq_prep_kernel<<<K_DIM / 8, 256, 0, stream>>>(cb, enorm, cbhi, cblo);

  const size_t shmem = (size_t)(2 * BUFELEMS + 2 * BROWS * CPAD) * 2; // ~203 KB < 320 KB/WGP
  vq_argmin_kernel<<<N_TOT / BROWS, 256, shmem, stream>>>(x, cbhi, cblo, enorm, amin);

  const int GBLK = 2048;
  vq_gather_loss_kernel<<<GBLK, 256, 0, stream>>>(x, cb, amin, out, partials);
  vq_loss_final_kernel<<<1, 256, 0, stream>>>(partials, GBLK, loss);
}
